// BitNetLinear_2156073582692
// MI455X (gfx1250) — matrix operand order repaired and checked by reading, not compiled, not run on hardware
//
#include <hip/hip_runtime.h>

// ---------------------------------------------------------------------------
// BitNet b1.58 linear for MI455X (gfx1250, wave32, WMMA)
// out[m,n] = scale * sum_k x[m,k] * q[n,k] + bias[n],  q in {-1,0,+1}
// M=8192, N=4096, K=4096.
// Fast path: prepack x->f16 and W->ternary f16 in ws (both L2-resident),
// then a pure f16 WMMA GEMM with ASYNCcnt-tracked global->LDS staging.
// Wave tile 64x64 (4x4 WMMA accs) -> 0.5 fragment loads per WMMA.
// ---------------------------------------------------------------------------

typedef __attribute__((ext_vector_type(16))) _Float16 v16h;
typedef __attribute__((ext_vector_type(8)))  _Float16 v8h;
typedef __attribute__((ext_vector_type(4)))  _Float16 h4;
typedef __attribute__((ext_vector_type(8)))  float    v8f;

#define THRESH     0.66666666f
#define K_DIM      4096
#define N_DIM      4096
#define BM         128
#define BN         256
#define BK         32
#define LDS_STRIDE 40          // 32 f16 payload + 8 f16 pad -> 80B row, conflict-free
#define NKT        (K_DIM / BK)

#define SHARED_AS __attribute__((address_space(3)))

// 16-byte per-lane global->LDS async copy (VFLAT GLOBAL_LOAD_ASYNC_TO_LDS_B128,
// tracked by ASYNCcnt). vdst = 32-bit LDS byte address, vaddr = 64-bit global.
__device__ __forceinline__ void copy16_to_lds(const _Float16* g, _Float16* l) {
    unsigned int lds_addr =
        (unsigned int)(unsigned long long)(SHARED_AS _Float16*)l;
    unsigned long long gaddr = (unsigned long long)g;
    asm volatile("global_load_async_to_lds_b128 %0, %1, off"
                 :: "v"(lds_addr), "v"(gaddr) : "memory");
}

__device__ __forceinline__ void wait_async_lds() {
    asm volatile("s_wait_asynccnt 0" ::: "memory");
}

// ------------------------- scale = clamp(mean(|W|)) -------------------------

__global__ void absmean_partial(const float* __restrict__ w,
                                float* __restrict__ partial, int n4) {
    __shared__ float sm[256];
    const float4* wv = (const float4*)w;
    float s = 0.0f;
    for (int i = blockIdx.x * blockDim.x + threadIdx.x; i < n4;
         i += gridDim.x * blockDim.x) {
        float4 v = wv[i];
        s += fabsf(v.x) + fabsf(v.y) + fabsf(v.z) + fabsf(v.w);
    }
    sm[threadIdx.x] = s;
    __syncthreads();
    for (int off = 128; off > 0; off >>= 1) {
        if (threadIdx.x < off) sm[threadIdx.x] += sm[threadIdx.x + off];
        __syncthreads();
    }
    if (threadIdx.x == 0) partial[blockIdx.x] = sm[0];
}

__global__ void absmean_final(const float* __restrict__ partial,
                              float* __restrict__ scale_out,
                              int nparts, float inv_count) {
    __shared__ float sm[256];
    float s = 0.0f;
    for (int i = threadIdx.x; i < nparts; i += 256) s += partial[i];
    sm[threadIdx.x] = s;
    __syncthreads();
    for (int off = 128; off > 0; off >>= 1) {
        if (threadIdx.x < off) sm[threadIdx.x] += sm[threadIdx.x + off];
        __syncthreads();
    }
    if (threadIdx.x == 0) {
        float m = sm[0] * inv_count;
        m = fminf(fmaxf(m, 1e-5f), 1000.0f);
        scale_out[0] = m;
    }
}

// --------------------------- quantize / convert ----------------------------

__device__ __forceinline__ float ternary_f32(float v, float inv_scale) {
    float wn = v * inv_scale;
    // wn > T -> +1 ; wn < -T -> -1 ; else (incl. NaN) -> 0
    return (fabsf(wn) > THRESH) ? copysignf(1.0f, wn) : 0.0f;
}

__global__ void quantize_w_f16(const float* __restrict__ w,
                               _Float16* __restrict__ qw,
                               const float* __restrict__ scale_p, int n4) {
    const float inv_s = 1.0f / scale_p[0];
    const float4* wv = (const float4*)w;
    h4* qv = (h4*)qw;
    for (int i = blockIdx.x * blockDim.x + threadIdx.x; i < n4;
         i += gridDim.x * blockDim.x) {
        float4 v = wv[i];
        h4 q = { (_Float16)ternary_f32(v.x, inv_s), (_Float16)ternary_f32(v.y, inv_s),
                 (_Float16)ternary_f32(v.z, inv_s), (_Float16)ternary_f32(v.w, inv_s) };
        qv[i] = q;
    }
}

__global__ void convert_x_f16(const float* __restrict__ x,
                              _Float16* __restrict__ xh, int n4) {
    const float4* xv = (const float4*)x;
    h4* hv = (h4*)xh;
    for (int i = blockIdx.x * blockDim.x + threadIdx.x; i < n4;
         i += gridDim.x * blockDim.x) {
        float4 v = xv[i];
        h4 h = { (_Float16)v.x, (_Float16)v.y, (_Float16)v.z, (_Float16)v.w };
        hv[i] = h;
    }
}

// ------------------------------ GEMM helpers -------------------------------

// A fragment (16x32 f16, lane = M row). Per CDNA5 ISA:
// lanes 0-15: regs0-3 = K[0..7],  regs4-7 = K[16..23]
// lanes16-31: regs0-3 = K[8..15], regs4-7 = K[24..31]
__device__ __forceinline__ v16h load_frag_a(const _Float16* base, int row, int khalf) {
    const _Float16* p = base + row * LDS_STRIDE + khalf * 8;
    union { v16h v; v8h h[2]; } u;
    u.h[0] = *(const v8h*)(p);       // ds_load_b128
    u.h[1] = *(const v8h*)(p + 16);  // ds_load_b128
    return u.v;
}

// B fragment (32x16 f16, lane%16 = N col; column n of B == row n of W).
// lanes 0-15: K[0..7], K[16..23]; lanes 16-31: K[8..15], K[24..31].
__device__ __forceinline__ v16h load_frag_b(const _Float16* base, int row, int khalf) {
    const _Float16* p = base + row * LDS_STRIDE + khalf * 8;
    union { v16h v; v8h h[2]; } u;
    u.h[0] = *(const v8h*)(p);
    u.h[1] = *(const v8h*)(p + 16);
    return u.v;
}

// --------------------- fast GEMM (prepacked f16 inputs) --------------------

__global__ void __launch_bounds__(256)
bitnet_gemm_f16(const _Float16* __restrict__ xh,   // [M, K] f16
                const _Float16* __restrict__ qw,   // [N, K] ternary f16
                const float* __restrict__ bias,    // [N]
                const float* __restrict__ scale_p,
                float* __restrict__ out) {         // [M, N] f32
    __shared__ __align__(16) _Float16 As[2][BM * LDS_STRIDE];
    __shared__ __align__(16) _Float16 Bs[2][BN * LDS_STRIDE];

    const int tid   = threadIdx.x;
    const int lane  = tid & 31;
    const int wave  = tid >> 5;       // 0..7
    const int wm    = wave & 1;       // 2 waves along M (64 rows each)
    const int wn    = wave >> 1;      // 4 waves along N (64 cols each)
    const int m0    = blockIdx.y * BM;
    const int n0    = blockIdx.x * BN;
    const int khalf = lane >> 4;
    const int lrow  = lane & 15;

    const float scale = scale_p[0];

    v8f acc[4][4];
#pragma unroll
    for (int mt = 0; mt < 4; ++mt)
#pragma unroll
        for (int nt = 0; nt < 4; ++nt) acc[mt][nt] = (v8f)(0.0f);

    // A tile = 128x32 f16 = 512 chunks of 8 f16  (2 per thread)
    // B tile = 256x32 f16 = 1024 chunks of 8 f16 (4 per thread)
    auto issue_tile = [&](int buf, int k0) {
#pragma unroll
        for (int i = 0; i < 2; ++i) {
            const int f = tid + i * 256;
            const int row = f >> 2, c8 = f & 3;
            copy16_to_lds(xh + (size_t)(m0 + row) * K_DIM + k0 + c8 * 8,
                          &As[buf][row * LDS_STRIDE + c8 * 8]);
        }
#pragma unroll
        for (int i = 0; i < 4; ++i) {
            const int f = tid + i * 256;
            const int row = f >> 2, c8 = f & 3;
            copy16_to_lds(qw + (size_t)(n0 + row) * K_DIM + k0 + c8 * 8,
                          &Bs[buf][row * LDS_STRIDE + c8 * 8]);
        }
    };

    issue_tile(0, 0);
    wait_async_lds();
    __syncthreads();

    for (int kt = 0; kt < NKT; ++kt) {
        const int cur = kt & 1;

        // Start async staging of the next K-tile into the other buffer.
        if (kt + 1 < NKT) issue_tile(cur ^ 1, (kt + 1) * BK);

        // Compute on current buffer: 4x4 WMMAs of 16x16x32.
        const _Float16* Ab = As[cur];
        const _Float16* Bb = Bs[cur];
        v16h af[4], bf[4];
#pragma unroll
        for (int mt = 0; mt < 4; ++mt)
            af[mt] = load_frag_a(Ab, wm * 64 + mt * 16 + lrow, khalf);
#pragma unroll
        for (int nt = 0; nt < 4; ++nt)
            bf[nt] = load_frag_b(Bb, wn * 64 + nt * 16 + lrow, khalf);
#pragma unroll
        for (int mt = 0; mt < 4; ++mt)
#pragma unroll
            for (int nt = 0; nt < 4; ++nt)
                acc[mt][nt] = __builtin_amdgcn_wmma_f32_16x16x32_f16(
                    false, af[mt], false, bf[nt],
                    (short)0, acc[mt][nt], false, false);

        if (kt + 1 < NKT) wait_async_lds();
        __syncthreads();
    }

    // Epilogue: C/D layout = 8 f32 VGPRs; lanes0-15: M=j, lanes16-31: M=j+8.
    const int mbase = m0 + wm * 64;
    const int nbase = n0 + wn * 64;
#pragma unroll
    for (int mt = 0; mt < 4; ++mt) {
#pragma unroll
        for (int nt = 0; nt < 4; ++nt) {
            const int col = nbase + nt * 16 + lrow;
            const float bv = bias[col];
#pragma unroll
            for (int j = 0; j < 8; ++j) {
                const int row = mbase + mt * 16 + khalf * 8 + j;
                out[(size_t)row * N_DIM + col] = acc[mt][nt][j] * scale + bv;
            }
        }
    }
}

// --------------- fallback GEMM (on-the-fly convert, small ws) --------------

#define FBM 128
#define FBN 128

__global__ void __launch_bounds__(256)
bitnet_gemm_fused(const float* __restrict__ x,      // [M, K] f32
                  const float* __restrict__ w,      // [N, K] f32
                  const float* __restrict__ bias,   // [N]
                  const float* __restrict__ scale_p,
                  float* __restrict__ out) {        // [M, N] f32
    __shared__ __align__(16) _Float16 As[2][FBM * LDS_STRIDE];
    __shared__ __align__(16) _Float16 Bs[2][FBN * LDS_STRIDE];

    const int tid   = threadIdx.x;
    const int lane  = tid & 31;
    const int wave  = tid >> 5;
    const int wm    = wave & 1;
    const int wn    = wave >> 1;
    const int m0    = blockIdx.y * FBM;
    const int n0    = blockIdx.x * FBN;
    const int khalf = lane >> 4;
    const int lrow  = lane & 15;

    const float scale = scale_p[0];
    const float inv_s = 1.0f / scale;

    v8f acc[4][2];
#pragma unroll
    for (int mt = 0; mt < 4; ++mt)
#pragma unroll
        for (int nt = 0; nt < 2; ++nt) acc[mt][nt] = (v8f)(0.0f);

    float4 ra[4], rb[4];
    auto load_stage = [&](int k0) {
#pragma unroll
        for (int i = 0; i < 4; ++i) {
            int f = tid + i * 256, row = f >> 3, c4 = f & 7;
            ra[i] = *(const float4*)(x + (size_t)(m0 + row) * K_DIM + k0 + c4 * 4);
            rb[i] = *(const float4*)(w + (size_t)(n0 + row) * K_DIM + k0 + c4 * 4);
        }
    };
    auto store_stage = [&](int buf) {
#pragma unroll
        for (int i = 0; i < 4; ++i) {
            int f = tid + i * 256, row = f >> 3, c4 = f & 7;
            h4 ah = { (_Float16)ra[i].x, (_Float16)ra[i].y,
                      (_Float16)ra[i].z, (_Float16)ra[i].w };
            h4 bh = { (_Float16)ternary_f32(rb[i].x, inv_s),
                      (_Float16)ternary_f32(rb[i].y, inv_s),
                      (_Float16)ternary_f32(rb[i].z, inv_s),
                      (_Float16)ternary_f32(rb[i].w, inv_s) };
            *(h4*)(&As[buf][row * LDS_STRIDE + c4 * 4]) = ah;
            *(h4*)(&Bs[buf][row * LDS_STRIDE + c4 * 4]) = bh;
        }
    };

    load_stage(0);
    store_stage(0);
    __syncthreads();

    for (int kt = 0; kt < NKT; ++kt) {
        const int cur = kt & 1;
        if (kt + 1 < NKT) load_stage((kt + 1) * BK);

        const _Float16* Ab = As[cur];
        const _Float16* Bb = Bs[cur];
        v16h af[4], bf[2];
#pragma unroll
        for (int mt = 0; mt < 4; ++mt)
            af[mt] = load_frag_a(Ab, wm * 64 + mt * 16 + lrow, khalf);
#pragma unroll
        for (int nt = 0; nt < 2; ++nt)
            bf[nt] = load_frag_b(Bb, wn * 32 + nt * 16 + lrow, khalf);
#pragma unroll
        for (int mt = 0; mt < 4; ++mt)
#pragma unroll
            for (int nt = 0; nt < 2; ++nt)
                acc[mt][nt] = __builtin_amdgcn_wmma_f32_16x16x32_f16(
                    false, af[mt], false, bf[nt],
                    (short)0, acc[mt][nt], false, false);

        if (kt + 1 < NKT) store_stage(cur ^ 1);
        __syncthreads();
    }

    const int mbase = m0 + wm * 64;
    const int nbase = n0 + wn * 32;
#pragma unroll
    for (int mt = 0; mt < 4; ++mt) {
#pragma unroll
        for (int nt = 0; nt < 2; ++nt) {
            const int col = nbase + nt * 16 + lrow;
            const float bv = bias[col];
#pragma unroll
            for (int j = 0; j < 8; ++j) {
                const int row = mbase + mt * 16 + khalf * 8 + j;
                out[(size_t)row * N_DIM + col] = acc[mt][nt][j] * scale + bv;
            }
        }
    }
}

// ------------------------------- launcher ----------------------------------

extern "C" void kernel_launch(void* const* d_in, const int* in_sizes, int n_in,
                              void* d_out, int out_size, void* d_ws, size_t ws_size,
                              hipStream_t stream) {
    const float* x    = (const float*)d_in[0];   // [B,S,K] f32
    const float* w    = (const float*)d_in[1];   // [N,K]   f32
    const float* bias = (const float*)d_in[2];   // [N]     f32
    float* out = (float*)d_out;

    const int nW = in_sizes[1];                  // N*K
    const int nX = in_sizes[0];                  // M*K
    const int M  = nX / K_DIM;                   // 8192

    const size_t qw_bytes  = (size_t)nW * sizeof(_Float16);
    const size_t xh_bytes  = (size_t)nX * sizeof(_Float16);
    const size_t red_bytes = (size_t)(1 + 1024) * sizeof(float);

    char* wsb = (char*)d_ws;

    if (ws_size >= qw_bytes + xh_bytes + red_bytes) {
        // Fast path: prepack both operands as f16 (L2-resident working set).
        _Float16* qw    = (_Float16*)wsb;
        _Float16* xh    = (_Float16*)(wsb + qw_bytes);
        float* scale    = (float*)(wsb + qw_bytes + xh_bytes);
        float* partials = scale + 1;

        absmean_partial<<<1024, 256, 0, stream>>>(w, partials, nW / 4);
        absmean_final<<<1, 256, 0, stream>>>(partials, scale, 1024, 1.0f / (float)nW);
        quantize_w_f16<<<4096, 256, 0, stream>>>(w, qw, scale, nW / 4);
        convert_x_f16<<<4096, 256, 0, stream>>>(x, xh, nX / 4);
        dim3 grid(N_DIM / BN, M / BM);           // (16, 64)
        bitnet_gemm_f16<<<grid, 256, 0, stream>>>(xh, qw, bias, scale, out);
    } else {
        // Fallback: on-the-fly convert/quantize inside the GEMM.
        float* scale    = (float*)wsb;
        float* partials = scale + 1;

        absmean_partial<<<1024, 256, 0, stream>>>(w, partials, nW / 4);
        absmean_final<<<1, 256, 0, stream>>>(partials, scale, 1024, 1.0f / (float)nW);
        dim3 grid(N_DIM / FBN, M / FBM);         // (32, 64)
        bitnet_gemm_fused<<<grid, 256, 0, stream>>>(x, w, bias, scale, out);
    }
}
